// RNN_21878563406421
// MI455X (gfx1250) — matrix operand order repaired and checked by reading, not compiled, not run on hardware
//
#include <hip/hip_runtime.h>
#include <hip/hip_bf16.h>
#include <math.h>

#define EMBD 1024
#define HIDD 1024
#define VOCAB 10000
#define SEQL 128
#define BATCH 64

#define BROW 40                 // padded halves per staged column (80B stride -> conflict-free banks)
#define BUFH (64 * BROW)        // halves per LDS buffer
#define BUFBYTES (BUFH * 2)

typedef unsigned short u16;
typedef u16    u16x8   __attribute__((ext_vector_type(8)));
typedef float  f32x8   __attribute__((ext_vector_type(8)));
typedef __bf16 bf16x16 __attribute__((ext_vector_type(16)));
typedef float  v8f     __attribute__((ext_vector_type(8)));

union FragU { u16x8 h[2]; bf16x16 v; };

__device__ __forceinline__ u16 f32_to_bf16_rne(float f) {
  union { float f; unsigned u; } x; x.f = f;
  unsigned u = x.u;
  unsigned r = u + 0x7FFFu + ((u >> 16) & 1u);
  return (u16)(r >> 16);
}
__device__ __forceinline__ float bf16_to_f32(u16 h) {
  union { unsigned u; float f; } x; x.u = ((unsigned)h) << 16;
  return x.f;
}

// ---- async global -> LDS staging (16B per lane per instruction, ASYNCcnt) ----
__device__ __forceinline__ void async_b128(unsigned lds_byte_off, const void* gaddr) {
  asm volatile("global_load_async_to_lds_b128 %0, %1, off"
               :: "v"(lds_byte_off), "v"((unsigned long long)(size_t)gaddr)
               : "memory");
}

// A fragment (16x32 bf16, MxK) per ISA layout:
// lane L: row m = m_base + (L&15); kg = (L>>4)*8
// elems 0..7 <- K=k0+kg+(0..7); elems 8..15 <- K=k0+16+kg+(0..7)
__device__ __forceinline__ bf16x16 load_a_bf16(const u16* __restrict__ A, int lda,
                                               int m_base, int k0, int lane) {
  int m  = m_base + (lane & 15);
  int kg = (lane >> 4) * 8;
  const u16* p = A + m * lda + k0 + kg;
  FragU f;
  f.h[0] = *(const u16x8*)(p);
  f.h[1] = *(const u16x8*)(p + 16);
  return f.v;
}

// Fallback: gather an fp32 embedding row and convert to bf16 on the fly.
__device__ __forceinline__ bf16x16 load_a_gather_f32(const float* __restrict__ row,
                                                     int k0, int lane) {
  int kg = (lane >> 4) * 8;
  f32x8 a = *(const f32x8*)(row + k0 + kg);
  f32x8 b = *(const f32x8*)(row + k0 + 16 + kg);
  FragU f;
  #pragma unroll
  for (int j = 0; j < 8; ++j) {
    f.h[0][j] = f32_to_bf16_rne(a[j]);
    f.h[1][j] = f32_to_bf16_rne(b[j]);
  }
  return f.v;
}

#define WMMA_BF16(acc, a, b) \
  (acc) = __builtin_amdgcn_wmma_f32_16x16x32_bf16(false, (a), false, (b), (short)0, (acc), false, false)

// ---------------------------------------------------------------------------
// Block-cooperative GEMM inner loop with LDS double-buffered B staging.
// Block = 128 threads (4 waves). Wave w computes rows [16w,16w+16); block owns
// 64 output columns starting at n0. B tile (64 cols x 32 K bf16 = 4KB) staged
// once per block per K-step via async copies; each wave reads fragments from
// LDS (ds_load_b128 pairs, conflict-free 80B column stride).
// A comes from bf16 buffer Abf (or fp32 gather row when Abf == nullptr).
// ---------------------------------------------------------------------------
__device__ __forceinline__ void gemm_lds(
    const u16* __restrict__ Abf, const float* __restrict__ xrow,
    const u16* __restrict__ W, int ldw, int K,
    int n0, int Nclamp, u16* Bsh, v8f* acc, int tid)
{
  const int lane   = tid & 31;
  const int m_base = (tid >> 5) * 16;
  const int scol   = tid >> 1;          // 0..63: staged column
  const int shalf  = tid & 1;           // 16-half chunk within the 32-K row
  const unsigned ldsBase = (unsigned)(size_t)Bsh;
  const unsigned colOff  = (unsigned)((scol * BROW + shalf * 16) * 2);
  int gcol = n0 + scol; if (gcol >= Nclamp) gcol = Nclamp - 1;   // clamp OOB tiles (stores guarded)
  const u16* gsrc = W + (size_t)gcol * ldw + shalf * 16;

  const int nsteps = K / 32;
  // prologue: stage K-step 0 into buffer 0
  async_b128(ldsBase + colOff,      gsrc);
  async_b128(ldsBase + colOff + 16, gsrc + 8);

  for (int i = 0; i < nsteps; ++i) {
    const u16* Bcur = Bsh + (i & 1) * BUFH;
    if (i + 1 < nsteps) {
      const u16* gn = gsrc + (i + 1) * 32;
      unsigned ln = ldsBase + (((i + 1) & 1) ? BUFBYTES : 0u) + colOff;
      async_b128(ln,      gn);
      async_b128(ln + 16, gn + 8);
      // in-order completion: <=2 outstanding means current buffer has landed
      asm volatile("s_wait_asynccnt 0x2" ::: "memory");
    } else {
      asm volatile("s_wait_asynccnt 0x0" ::: "memory");
    }
    __syncthreads();

    const int k0 = i * 32;
    bf16x16 a = Abf ? load_a_bf16(Abf, K, m_base, k0, lane)
                    : load_a_gather_f32(xrow, k0, lane);
    #pragma unroll
    for (int s = 0; s < 4; ++s) {
      // B fragment: lane col n = s*16 + (lane&15), kg = (lane>>4)*8; elems 0..7 <- K=kg+(0..7), 8..15 <- K=16+kg+(0..7)
      const u16* p = Bcur + (s * 16 + (lane & 15)) * BROW + (lane >> 4) * 8;
      FragU f;
      f.h[0] = *(const u16x8*)(p);
      f.h[1] = *(const u16x8*)(p + 16);
      WMMA_BF16(acc[s], a, f.v);
    }
    __syncthreads();   // protect buffer reuse before next prefetch overwrites
  }
}

// C/D layout: lane holds one output column n; VGPR r -> row m_base + (lane>>4)*8 + r.
__device__ __forceinline__ void store_tile_bf16(u16* __restrict__ out, int ldo,
                                                int m_base, int n, const float* __restrict__ bias,
                                                bool do_tanh, v8f acc, int lane) {
  float bv = bias[n];
  int moff = m_base + (lane >> 4) * 8;
  #pragma unroll
  for (int r = 0; r < 8; ++r) {
    float v = acc[r] + bv;
    if (do_tanh) v = tanhf(v);
    out[(moff + r) * ldo + n] = f32_to_bf16_rne(v);
  }
}

__device__ __forceinline__ void store_tile_f32(float* __restrict__ out, int ldo,
                                               int m_base, int n, const float* __restrict__ bias,
                                               v8f acc, int lane) {
  float bv = bias[n];
  int moff = m_base + (lane >> 4) * 8;
  #pragma unroll
  for (int r = 0; r < 8; ++r)
    out[(size_t)(moff + r) * ldo + n] = acc[r] + bv;
}

// ---------------------------------------------------------------------------
// Step kernel 1: jobs via blockIdx.y
//   job 0: y0     = h0 @ Wy0^T + by0                      (N=1024, bf16 out)
//   job 1: h0n    = tanh(x_t @ Wx0^T + h0 @ Wh0^T + bh0)  (N=1024, bf16 out)
//   job 2: logits = h1 @ Wy1^T + by1                      (N=10000, fp32 out)
// ---------------------------------------------------------------------------
__global__ __launch_bounds__(128, 1) void rnn_step1(
    const u16* __restrict__ h0, const u16* __restrict__ h1,
    const u16* __restrict__ Wy0, const u16* __restrict__ Wx0,
    const u16* __restrict__ Wh0, const u16* __restrict__ Wy1,
    const float* __restrict__ by0, const float* __restrict__ bh0,
    const float* __restrict__ by1,
    const u16* __restrict__ x_t,      // precomputed bf16 x_t, or nullptr
    const int* __restrict__ tok_t, const float* __restrict__ emb,
    u16* __restrict__ y0_out, u16* __restrict__ h0_out,
    float* __restrict__ logits_t)
{
  __shared__ u16 Bsh[2 * BUFH];
  const int tid    = threadIdx.x;
  const int lane   = tid & 31;
  const int m_base = (tid >> 5) * 16;
  const int job    = blockIdx.y;
  const int nb     = blockIdx.x;

  v8f acc[4] = {};

  if (job == 0) {
    if (nb >= HIDD / 64) return;
    gemm_lds(h0, nullptr, Wy0, HIDD, HIDD, nb * 64, HIDD, Bsh, acc, tid);
    #pragma unroll
    for (int s = 0; s < 4; ++s)
      store_tile_bf16(y0_out, HIDD, m_base, nb * 64 + s * 16 + (lane & 15), by0, false, acc[s], lane);
  } else if (job == 1) {
    if (nb >= HIDD / 64) return;
    if (x_t) {
      gemm_lds(x_t, nullptr, Wx0, EMBD, EMBD, nb * 64, HIDD, Bsh, acc, tid);
    } else {
      const float* xrow = emb + (size_t)tok_t[m_base + (lane & 15)] * EMBD;
      gemm_lds(nullptr, xrow, Wx0, EMBD, EMBD, nb * 64, HIDD, Bsh, acc, tid);
    }
    gemm_lds(h0, nullptr, Wh0, HIDD, HIDD, nb * 64, HIDD, Bsh, acc, tid);
    #pragma unroll
    for (int s = 0; s < 4; ++s)
      store_tile_bf16(h0_out, HIDD, m_base, nb * 64 + s * 16 + (lane & 15), bh0, true, acc[s], lane);
  } else {
    gemm_lds(h1, nullptr, Wy1, HIDD, HIDD, nb * 64, VOCAB, Bsh, acc, tid);
    #pragma unroll
    for (int s = 0; s < 4; ++s) {
      int n_base = nb * 64 + s * 16;
      if (n_base >= VOCAB) continue;   // uniform; VOCAB % 16 == 0 -> tiles fully in/out
      store_tile_f32(logits_t, VOCAB, m_base, n_base + (lane & 15), by1, acc[s], lane);
    }
  }
}

// Step kernel 2: h1n = tanh(y0 @ Wx1^T + h1 @ Wh1^T + bh1)   (N=1024)
__global__ __launch_bounds__(128, 1) void rnn_step2(
    const u16* __restrict__ y0, const u16* __restrict__ h1,
    const u16* __restrict__ Wx1, const u16* __restrict__ Wh1,
    const float* __restrict__ bh1, u16* __restrict__ h1_out)
{
  __shared__ u16 Bsh[2 * BUFH];
  const int tid    = threadIdx.x;
  const int lane   = tid & 31;
  const int m_base = (tid >> 5) * 16;
  const int nb     = blockIdx.x;

  v8f acc[4] = {};
  gemm_lds(y0, nullptr, Wx1, HIDD, HIDD, nb * 64, HIDD, Bsh, acc, tid);
  gemm_lds(h1, nullptr, Wh1, HIDD, HIDD, nb * 64, HIDD, Bsh, acc, tid);
  #pragma unroll
  for (int s = 0; s < 4; ++s)
    store_tile_bf16(h1_out, HIDD, m_base, nb * 64 + s * 16 + (lane & 15), bh1, true, acc[s], lane);
}

// ---------------------------------------------------------------------------
// Prologue / epilogue kernels
// ---------------------------------------------------------------------------
__global__ void cvt_f32_bf16(const float* __restrict__ s, u16* __restrict__ d, int n) {
  int i = blockIdx.x * blockDim.x + threadIdx.x;
  int stride = gridDim.x * blockDim.x;
  for (; i < n; i += stride) d[i] = f32_to_bf16_rne(s[i]);
}

// Materialize all embedded inputs in bf16: xall[t][b][:] = bf16(emb[tok[t][b]][:])
__global__ void embed_all(const int* __restrict__ toks, const float* __restrict__ emb,
                          u16* __restrict__ xall) {
  int tb = blockIdx.x;                          // t*BATCH + b
  const float* src = emb + (size_t)toks[tb] * EMBD;
  u16* dst = xall + (size_t)tb * EMBD;
  for (int e = threadIdx.x; e < EMBD; e += blockDim.x)
    dst[e] = f32_to_bf16_rne(src[e]);
}

__global__ void init_hidden(const float* __restrict__ h, u16* __restrict__ h0, u16* __restrict__ h1) {
  int i = blockIdx.x * blockDim.x + threadIdx.x;   // 0..BATCH*HIDD-1
  h0[i] = f32_to_bf16_rne(h[i]);
  h1[i] = f32_to_bf16_rne(h[BATCH * HIDD + i]);
}

__global__ void finalize_hidden(const u16* __restrict__ h0, const u16* __restrict__ h1,
                                float* __restrict__ o) {
  int i = blockIdx.x * blockDim.x + threadIdx.x;   // 0..BATCH*HIDD-1
  o[i] = bf16_to_f32(h0[i]);
  o[BATCH * HIDD + i] = bf16_to_f32(h1[i]);
}

// ---------------------------------------------------------------------------
extern "C" void kernel_launch(void* const* d_in, const int* in_sizes, int n_in,
                              void* d_out, int out_size, void* d_ws, size_t ws_size,
                              hipStream_t stream) {
  (void)in_sizes; (void)n_in; (void)out_size;

  const int*   inputs = (const int*)d_in[0];
  const float* hidden = (const float*)d_in[1];
  const float* emb    = (const float*)d_in[2];
  const float* Wx0f   = (const float*)d_in[3];
  const float* Wh0f   = (const float*)d_in[4];
  const float* bh0    = (const float*)d_in[5];
  const float* Wy0f   = (const float*)d_in[6];
  const float* by0    = (const float*)d_in[7];
  const float* Wx1f   = (const float*)d_in[8];
  const float* Wh1f   = (const float*)d_in[9];
  const float* bh1    = (const float*)d_in[10];
  const float* Wy1f   = (const float*)d_in[11];
  const float* by1    = (const float*)d_in[12];

  float* out = (float*)d_out;

  // workspace carve-up (bf16 weights stay L2-resident: ~30 MB total)
  char* w = (char*)d_ws;
  const size_t W1M = (size_t)HIDD * HIDD * sizeof(u16);      // 2 MB
  u16* Wy0b = (u16*)w; w += W1M;
  u16* Wx0b = (u16*)w; w += W1M;
  u16* Wh0b = (u16*)w; w += W1M;
  u16* Wx1b = (u16*)w; w += W1M;
  u16* Wh1b = (u16*)w; w += W1M;
  u16* Wy1b = (u16*)w; w += (size_t)VOCAB * HIDD * sizeof(u16);
  const size_t HS = (size_t)BATCH * HIDD * sizeof(u16);
  u16* h0b[2]; h0b[0] = (u16*)w; w += HS; h0b[1] = (u16*)w; w += HS;
  u16* h1b[2]; h1b[0] = (u16*)w; w += HS; h1b[1] = (u16*)w; w += HS;
  u16* y0b = (u16*)w; w += HS;
  // optional: all-timestep bf16 embeddings (16.8 MB); fall back to in-kernel gather
  const size_t embBytes = (size_t)SEQL * BATCH * EMBD * sizeof(u16);
  size_t used = (size_t)(w - (char*)d_ws);
  u16* xall = (ws_size >= used + embBytes) ? (u16*)w : nullptr;

  // prologue: one-shot fp32 -> bf16 weight conversion + hidden init (+ embeddings)
  cvt_f32_bf16<<<1024, 256, 0, stream>>>(Wy0f, Wy0b, HIDD * HIDD);
  cvt_f32_bf16<<<1024, 256, 0, stream>>>(Wx0f, Wx0b, HIDD * EMBD);
  cvt_f32_bf16<<<1024, 256, 0, stream>>>(Wh0f, Wh0b, HIDD * HIDD);
  cvt_f32_bf16<<<1024, 256, 0, stream>>>(Wx1f, Wx1b, HIDD * HIDD);
  cvt_f32_bf16<<<1024, 256, 0, stream>>>(Wh1f, Wh1b, HIDD * HIDD);
  cvt_f32_bf16<<<2048, 256, 0, stream>>>(Wy1f, Wy1b, VOCAB * HIDD);
  init_hidden<<<(BATCH * HIDD) / 256, 256, 0, stream>>>(hidden, h0b[0], h1b[0]);
  if (xall) embed_all<<<SEQL * BATCH, 256, 0, stream>>>(inputs, emb, xall);

  // sequential scan: 2 kernels per timestep, ping-pong hidden state in bf16
  const dim3 g1((VOCAB + 63) / 64, 3);   // jobs: y0 / h0n / logits
  const dim3 g2(HIDD / 64, 1);
  for (int t = 0; t < SEQL; ++t) {
    const u16* h0i = h0b[t & 1];  u16* h0o = h0b[(t + 1) & 1];
    const u16* h1i = h1b[t & 1];  u16* h1o = h1b[(t + 1) & 1];
    rnn_step1<<<g1, 128, 0, stream>>>(h0i, h1i, Wy0b, Wx0b, Wh0b, Wy1b,
                                      by0, bh0, by1,
                                      xall ? xall + (size_t)t * BATCH * EMBD : (const u16*)nullptr,
                                      inputs + (size_t)t * BATCH, emb,
                                      y0b, h0o,
                                      out + (size_t)t * BATCH * VOCAB);
    rnn_step2<<<g2, 128, 0, stream>>>(y0b, h1i, Wx1b, Wh1b, bh1, h1o);
  }

  // epilogue: hidden_final (fp32) after logits block
  finalize_hidden<<<(BATCH * HIDD) / 256, 256, 0, stream>>>(
      h0b[SEQL & 1], h1b[SEQL & 1], out + (size_t)SEQL * BATCH * VOCAB);
}
